// EncoderBlock_34385508172204
// MI455X (gfx1250) — matrix operand order repaired and checked by reading, not compiled, not run on hardware
//
#include <hip/hip_runtime.h>
#include <math.h>

// ---------------- CDNA5 WMMA types ----------------
typedef __attribute__((ext_vector_type(16))) __bf16 v16bf;
typedef __attribute__((ext_vector_type(8)))  float  v8f;

#define D_MODEL 1024
#define N_HEAD  16
#define DKH     64
#define D_FF    4096
#define SEQ     1024
#define BATCH   8
#define MTOT    (BATCH * SEQ)

// fp32 -> bf16 bits, round-to-nearest-even (pure bit manipulation; no scalar bf16 ops)
static __device__ __forceinline__ unsigned short f2bf(float f) {
    union { float f; unsigned u; } in; in.f = f;
    unsigned u = in.u;
    u += 0x7fffu + ((u >> 16) & 1u);
    return (unsigned short)(u >> 16);
}

union ABPack { uint4 u[2]; v16bf v; };

// Build a 16-element bf16 WMMA operand from two contiguous 16-byte chunks.
static __device__ __forceinline__ v16bf load2x16(const unsigned short* p0,
                                                 const unsigned short* p1) {
    ABPack t;
    t.u[0] = *(const uint4*)p0;
    t.u[1] = *(const uint4*)p1;
    return t.v;
}

static __device__ __forceinline__ v8f zero8() {
    v8f z = {0.f, 0.f, 0.f, 0.f, 0.f, 0.f, 0.f, 0.f};
    return z;
}

// ---------------- weight fp32 [K,N] -> bf16 transposed [N,K] ----------------
__global__ void w_transpose_bf16(const float* __restrict__ W,
                                 unsigned short* __restrict__ Wt, int K, int N) {
    int i = blockIdx.x * blockDim.x + threadIdx.x;
    if (i >= K * N) return;
    int k = i / N, n = i % N;
    Wt[(size_t)n * K + k] = f2bf(W[i]);
}

// ---------------- LayerNorm (scalar alpha/beta), fp32 -> bf16 ----------------
__global__ __launch_bounds__(256)
void layernorm_bf16(const float* __restrict__ x, const float* __restrict__ alpha,
                    const float* __restrict__ beta, unsigned short* __restrict__ out) {
    const int row = blockIdx.x;
    const float* xr = x + (size_t)row * D_MODEL;
    float v[4], s = 0.f, ss = 0.f;
#pragma unroll
    for (int i = 0; i < 4; ++i) {
        v[i] = xr[threadIdx.x + i * 256];
        s += v[i]; ss += v[i] * v[i];
    }
#pragma unroll
    for (int m = 16; m >= 1; m >>= 1) {
        s  += __shfl_xor(s,  m, 32);
        ss += __shfl_xor(ss, m, 32);
    }
    __shared__ float sh[2][8];
    int wid = threadIdx.x >> 5;
    if ((threadIdx.x & 31) == 0) { sh[0][wid] = s; sh[1][wid] = ss; }
    __syncthreads();
    if (threadIdx.x == 0) {
        float ts = 0.f, tss = 0.f;
        for (int i = 0; i < 8; ++i) { ts += sh[0][i]; tss += sh[1][i]; }
        sh[0][0] = ts; sh[1][0] = tss;
    }
    __syncthreads();
    float mean = sh[0][0] * (1.f / D_MODEL);
    float var  = sh[1][0] * (1.f / D_MODEL) - mean * mean;
    float stdv = sqrtf(fmaxf(var, 0.f));
    float inv  = alpha[0] / (stdv + 1e-6f);
    float b    = beta[0];
    unsigned short* orow = out + (size_t)row * D_MODEL;
#pragma unroll
    for (int i = 0; i < 4; ++i)
        orow[threadIdx.x + i * 256] = f2bf((v[i] - mean) * inv + b);
}

// ---------------- WMMA GEMM: C = A[M,K] @ W  (Bt = W^T as [N,K] bf16) ----------
// Block tile 128x128 (8 waves as 4M x 2N), wave tile 32x64 (2 A-tiles x 4 B-tiles).
// out_mode: 0 = [M,N];  1 = q/k layout [B,H,S,64];  2 = v layout [B,H,64,S]
__global__ __launch_bounds__(256)
void gemm_bf16_wmma(const unsigned short* __restrict__ A,
                    const unsigned short* __restrict__ Bt,
                    const float* __restrict__ bias,
                    const float* __restrict__ residual,
                    float* __restrict__ outF, unsigned short* __restrict__ outB,
                    int M, int N, int K, int out_mode, int do_gelu) {
    const int tid  = threadIdx.x;
    const int lane = tid & 31;
    const int wid  = tid >> 5;
    const int l15  = lane & 15;
    const int half = (lane >> 4) & 1;
    const int wm = wid & 3, wn = wid >> 2;
    const int m_base = blockIdx.y * 128 + wm * 32;
    const int n_base = blockIdx.x * 128 + wn * 64;

    v8f acc[2][4];
#pragma unroll
    for (int i = 0; i < 2; ++i)
#pragma unroll
        for (int j = 0; j < 4; ++j) acc[i][j] = zero8();

    const unsigned short* arow[2];
    arow[0] = A + (size_t)(m_base + l15) * K;
    arow[1] = A + (size_t)(m_base + 16 + l15) * K;
    const unsigned short* brow[4];
#pragma unroll
    for (int j = 0; j < 4; ++j)
        brow[j] = Bt + (size_t)(n_base + j * 16 + l15) * K;

    for (int k0 = 0; k0 < K; k0 += 32) {
        // prefetch the A stream ahead (B = weights, resident in L2)
        __builtin_prefetch(arow[0] + k0 + 128, 0, 1);
        __builtin_prefetch(arow[1] + k0 + 128, 0, 1);
        v16bf a0 = load2x16(arow[0] + k0 + half * 8, arow[0] + k0 + 16 + half * 8);
        v16bf a1 = load2x16(arow[1] + k0 + half * 8, arow[1] + k0 + 16 + half * 8);
        v16bf b[4];
#pragma unroll
        for (int j = 0; j < 4; ++j)
            b[j] = load2x16(brow[j] + k0 + half * 8, brow[j] + k0 + 16 + half * 8);
#pragma unroll
        for (int j = 0; j < 4; ++j) {
            acc[0][j] = __builtin_amdgcn_wmma_f32_16x16x32_bf16(false, a0, false, b[j], (short)0, acc[0][j], false, false);
            acc[1][j] = __builtin_amdgcn_wmma_f32_16x16x32_bf16(false, a1, false, b[j], (short)0, acc[1][j], false, false);
        }
    }

#pragma unroll
    for (int mi = 0; mi < 2; ++mi)
#pragma unroll
        for (int nj = 0; nj < 4; ++nj) {
            int col = n_base + nj * 16 + l15;
            float bv = bias ? bias[col] : 0.f;
#pragma unroll
            for (int r = 0; r < 8; ++r) {
                int row = m_base + mi * 16 + half * 8 + r;
                float v = acc[mi][nj][r] + bv;
                if (do_gelu) v = 0.5f * v * (1.0f + erff(v * 0.70710678118654752f));
                if (residual) v += residual[(size_t)row * N + col];
                if (outF) outF[(size_t)row * N + col] = v;
                if (outB) {
                    size_t idx;
                    if (out_mode == 0) {
                        idx = (size_t)row * N + col;
                    } else {
                        int h = col >> 6, d = col & 63;
                        int b2 = row >> 10, s = row & 1023;
                        if (out_mode == 1)
                            idx = (((size_t)(b2 * N_HEAD + h) * SEQ) + s) * DKH + d;
                        else
                            idx = (((size_t)(b2 * N_HEAD + h) * DKH) + d) * SEQ + s;
                    }
                    outB[idx] = f2bf(v);
                }
            }
        }
}

// ---------------- Flash attention with WMMA ----------------
// Q,K: [B,H,S,64] bf16; Vt: [B,H,64,S] bf16; ctx: [B,S,D] bf16
__global__ __launch_bounds__(128)
void attention_wmma(const unsigned short* __restrict__ Q,
                    const unsigned short* __restrict__ Kb,
                    const unsigned short* __restrict__ Vt,
                    const int* __restrict__ mask,
                    unsigned short* __restrict__ ctx) {
    const int tid  = threadIdx.x;
    const int lane = tid & 31;
    const int wid  = tid >> 5;   // 0..3
    const int l15  = lane & 15;
    const int half = (lane >> 4) & 1;
    const int bh = blockIdx.y;   // 0..B*H-1
    const int b  = bh >> 4;
    const int h  = bh & 15;
    const int qbase = blockIdx.x * 64 + wid * 16;

    const unsigned short* Qh = Q  + (size_t)bh * SEQ * DKH;
    const unsigned short* Kh = Kb + (size_t)bh * SEQ * DKH;
    const unsigned short* Vh = Vt + (size_t)bh * DKH * SEQ;
    const int* mrow = mask + b * SEQ;

    const unsigned short* qrow = Qh + (size_t)(qbase + l15) * DKH;
    v16bf aQ0 = load2x16(qrow + half * 8,      qrow + 16 + half * 8);
    v16bf aQ1 = load2x16(qrow + 32 + half * 8, qrow + 48 + half * 8);

    float mrun[8], lrun[8];
    v8f o[4];
#pragma unroll
    for (int r = 0; r < 8; ++r) { mrun[r] = -3.0e38f; lrun[r] = 0.f; }
#pragma unroll
    for (int d = 0; d < 4; ++d) o[d] = zero8();

    __shared__ __align__(16) unsigned short pbuf[4][16][32];

    for (int key0 = 0; key0 < SEQ; key0 += 32) {
        v8f sc[2]; sc[0] = zero8(); sc[1] = zero8();
#pragma unroll
        for (int t = 0; t < 2; ++t) {
            const unsigned short* krow = Kh + (size_t)(key0 + t * 16 + l15) * DKH;
            v16bf bK0 = load2x16(krow + half * 8,      krow + 16 + half * 8);
            v16bf bK1 = load2x16(krow + 32 + half * 8, krow + 48 + half * 8);
            sc[t] = __builtin_amdgcn_wmma_f32_16x16x32_bf16(false, aQ0, false, bK0, (short)0, sc[t], false, false);
            sc[t] = __builtin_amdgcn_wmma_f32_16x16x32_bf16(false, aQ1, false, bK1, (short)0, sc[t], false, false);
        }
        // scale 1/sqrt(64) + mask
#pragma unroll
        for (int t = 0; t < 2; ++t) {
            float madd = (mrow[key0 + t * 16 + l15] == 0) ? -1.0e9f : 0.0f;
#pragma unroll
            for (int r = 0; r < 8; ++r) sc[t][r] = sc[t][r] * 0.125f + madd;
        }
        // streaming softmax per row (rows live across the 16 lanes of a half-wave)
#pragma unroll
        for (int r = 0; r < 8; ++r) {
            float mx = fmaxf(sc[0][r], sc[1][r]);
#pragma unroll
            for (int msk = 8; msk >= 1; msk >>= 1)
                mx = fmaxf(mx, __shfl_xor(mx, msk, 32));
            float mnew    = fmaxf(mrun[r], mx);
            float rescale = __expf(mrun[r] - mnew);
            float p0 = __expf(sc[0][r] - mnew);
            float p1 = __expf(sc[1][r] - mnew);
            sc[0][r] = p0; sc[1][r] = p1;
            float sum = p0 + p1;
#pragma unroll
            for (int msk = 8; msk >= 1; msk >>= 1)
                sum += __shfl_xor(sum, msk, 32);
            lrun[r] = lrun[r] * rescale + sum;
            mrun[r] = mnew;
#pragma unroll
            for (int d = 0; d < 4; ++d) o[d][r] *= rescale;
        }
        // C-layout -> A-layout via LDS (per-wave region)
#pragma unroll
        for (int r = 0; r < 8; ++r) {
            pbuf[wid][half * 8 + r][l15]      = f2bf(sc[0][r]);
            pbuf[wid][half * 8 + r][16 + l15] = f2bf(sc[1][r]);
        }
        __syncthreads();
        const unsigned short* prow = &pbuf[wid][l15][0];
        v16bf aP = load2x16(prow + half * 8, prow + 16 + half * 8);
        __syncthreads();
        // O += P @ V
#pragma unroll
        for (int d = 0; d < 4; ++d) {
            const unsigned short* vrow = Vh + (size_t)(d * 16 + l15) * SEQ + key0;
            v16bf bV = load2x16(vrow + half * 8, vrow + 16 + half * 8);
            o[d] = __builtin_amdgcn_wmma_f32_16x16x32_bf16(false, aP, false, bV, (short)0, o[d], false, false);
        }
    }
    // normalize and write ctx[b, s, h*64 + d]
#pragma unroll
    for (int d = 0; d < 4; ++d) {
#pragma unroll
        for (int r = 0; r < 8; ++r) {
            int srow = qbase + half * 8 + r;
            float v = o[d][r] / lrun[r];
            size_t idx = ((size_t)(b * SEQ + srow)) * D_MODEL + h * DKH + d * 16 + l15;
            ctx[idx] = f2bf(v);
        }
    }
}

// ---------------- host orchestration ----------------
extern "C" void kernel_launch(void* const* d_in, const int* in_sizes, int n_in,
                              void* d_out, int out_size, void* d_ws, size_t ws_size,
                              hipStream_t stream) {
    (void)in_sizes; (void)n_in; (void)out_size; (void)ws_size;
    const float* x    = (const float*)d_in[0];
    const int*   mask = (const int*)d_in[1];
    const float* wq = (const float*)d_in[2];  const float* bq = (const float*)d_in[3];
    const float* wk = (const float*)d_in[4];  const float* bk = (const float*)d_in[5];
    const float* wv = (const float*)d_in[6];  const float* bv = (const float*)d_in[7];
    const float* wo = (const float*)d_in[8];  const float* bo = (const float*)d_in[9];
    const float* w1 = (const float*)d_in[10]; const float* b1 = (const float*)d_in[11];
    const float* w2 = (const float*)d_in[12]; const float* b2 = (const float*)d_in[13];
    const float* a1 = (const float*)d_in[14]; const float* o1 = (const float*)d_in[15];
    const float* a2 = (const float*)d_in[16]; const float* o2 = (const float*)d_in[17];

    char* ws = (char*)d_ws;
    size_t off = 0;
    auto alloc = [&](size_t bytes) -> void* {
        void* p = ws + off;
        off += (bytes + 255) & ~(size_t)255;
        return p;
    };
    unsigned short* wq_t = (unsigned short*)alloc((size_t)D_MODEL * D_MODEL * 2);
    unsigned short* wk_t = (unsigned short*)alloc((size_t)D_MODEL * D_MODEL * 2);
    unsigned short* wv_t = (unsigned short*)alloc((size_t)D_MODEL * D_MODEL * 2);
    unsigned short* wo_t = (unsigned short*)alloc((size_t)D_MODEL * D_MODEL * 2);
    unsigned short* w1_t = (unsigned short*)alloc((size_t)D_MODEL * D_FF * 2);
    unsigned short* w2_t = (unsigned short*)alloc((size_t)D_FF * D_MODEL * 2);
    unsigned short* xn1  = (unsigned short*)alloc((size_t)MTOT * D_MODEL * 2);
    unsigned short* qb   = (unsigned short*)alloc((size_t)MTOT * D_MODEL * 2);
    unsigned short* kb   = (unsigned short*)alloc((size_t)MTOT * D_MODEL * 2);
    unsigned short* vtb  = (unsigned short*)alloc((size_t)MTOT * D_MODEL * 2);
    unsigned short* ctx  = (unsigned short*)alloc((size_t)MTOT * D_MODEL * 2);
    float*          x2   = (float*)alloc((size_t)MTOT * D_MODEL * 4);
    unsigned short* xn2  = (unsigned short*)alloc((size_t)MTOT * D_MODEL * 2);
    unsigned short* hb   = (unsigned short*)alloc((size_t)MTOT * D_FF * 2);

    // 1) weight convert + transpose to bf16 [N,K]
    int n_dd = (D_MODEL * D_MODEL + 255) / 256;
    int n_df = (D_MODEL * D_FF + 255) / 256;
    w_transpose_bf16<<<n_dd, 256, 0, stream>>>(wq, wq_t, D_MODEL, D_MODEL);
    w_transpose_bf16<<<n_dd, 256, 0, stream>>>(wk, wk_t, D_MODEL, D_MODEL);
    w_transpose_bf16<<<n_dd, 256, 0, stream>>>(wv, wv_t, D_MODEL, D_MODEL);
    w_transpose_bf16<<<n_dd, 256, 0, stream>>>(wo, wo_t, D_MODEL, D_MODEL);
    w_transpose_bf16<<<n_df, 256, 0, stream>>>(w1, w1_t, D_MODEL, D_FF);
    w_transpose_bf16<<<n_df, 256, 0, stream>>>(w2, w2_t, D_FF, D_MODEL);

    // 2) LN1
    layernorm_bf16<<<MTOT, 256, 0, stream>>>(x, a1, o1, xn1);

    // 3) Q/K/V projections (WMMA)
    dim3 gp(D_MODEL / 128, MTOT / 128);
    gemm_bf16_wmma<<<gp, 256, 0, stream>>>(xn1, wq_t, bq, nullptr, nullptr, qb,
                                           MTOT, D_MODEL, D_MODEL, 1, 0);
    gemm_bf16_wmma<<<gp, 256, 0, stream>>>(xn1, wk_t, bk, nullptr, nullptr, kb,
                                           MTOT, D_MODEL, D_MODEL, 1, 0);
    gemm_bf16_wmma<<<gp, 256, 0, stream>>>(xn1, wv_t, bv, nullptr, nullptr, vtb,
                                           MTOT, D_MODEL, D_MODEL, 2, 0);

    // 4) attention
    dim3 ga(SEQ / 64, BATCH * N_HEAD);
    attention_wmma<<<ga, 128, 0, stream>>>(qb, kb, vtb, mask, ctx);

    // 5) output projection + residual -> x2 (fp32)
    gemm_bf16_wmma<<<gp, 256, 0, stream>>>(ctx, wo_t, bo, x, x2, nullptr,
                                           MTOT, D_MODEL, D_MODEL, 0, 0);

    // 6) LN2
    layernorm_bf16<<<MTOT, 256, 0, stream>>>(x2, a2, o2, xn2);

    // 7) FFN1 + exact GELU -> bf16
    dim3 gf1(D_FF / 128, MTOT / 128);
    gemm_bf16_wmma<<<gf1, 256, 0, stream>>>(xn2, w1_t, b1, nullptr, nullptr, hb,
                                            MTOT, D_FF, D_MODEL, 0, 1);

    // 8) FFN2 + residual -> d_out (fp32)
    gemm_bf16_wmma<<<gp, 256, 0, stream>>>(hb, w2_t, b2, x2, (float*)d_out, nullptr,
                                           MTOT, D_MODEL, D_FF, 0, 0);
}
